// PointNetSetAbstraction_50972671869076
// MI455X (gfx1250) — matrix operand order repaired and checked by reading, not compiled, not run on hardware
//
#include <hip/hip_runtime.h>

typedef __attribute__((ext_vector_type(16))) _Float16 v16h;
typedef __attribute__((ext_vector_type(8)))  float    v8f;

#define B_      16
#define N_      4096
#define NPOINT_ 1024
#define NSAMPLE_ 32
#define P_      (B_*NPOINT_*NSAMPLE_)   /* 524288 positions (GEMM M) */
#define MT_     (P_/16)                 /* 32768 M-tiles */
#define RAD2    0.04f                   /* 0.2^2 */
#define EPS_    1e-5f

// ---- WMMA fragment K maps (CDNA5 ISA 7.12.2, 16-bit 16x16x32) ----
// A 16x32: lanes 0-15 rows M=0..15 hold K {0..7,16..23}; lanes 16-31 hold K {8..15,24..31}
__device__ __forceinline__ int kmapA(int i, int half){ return (i < 8 ? i : i + 8) + 8*half; }
// B 32x16 contiguous map K = i + 16*half (unused; bsw_kernel packs B with kmapA)
__device__ __forceinline__ int kmapB(int i, int half){ return i + 16*half; }

// Swizzled A storage: [Mtile][kchunk][lane][i] contiguous -> one 32B load per lane
__device__ __forceinline__ size_t aswz(size_t p, int k, int numK){
  size_t mtile = p >> 4; int r = (int)(p & 15);
  int kc = k >> 5, kk = k & 31;
  int half = (kk >> 3) & 1;
  int t = kk - 8*half;
  int i = (t < 8) ? t : (t - 8);
  int lane = half*16 + r;
  return ((mtile*(size_t)numK + kc)*32 + (size_t)lane)*16 + (size_t)i;
}

// ================= FPS: one workgroup per batch, xyz in LDS =================
__global__ void fps_kernel(const float* __restrict__ xyz, float* __restrict__ out){
  __shared__ float sx[N_*3];
  __shared__ float rd[256];
  __shared__ int   ri[256];
  int b = blockIdx.x, tid = threadIdx.x;
  const float* gx = xyz + (size_t)b*N_*3;
  for (int i = tid; i < N_*3; i += 256) sx[i] = gx[i];
  float dist[16];
#pragma unroll
  for (int r = 0; r < 16; ++r) dist[r] = 1e10f;
  __syncthreads();
  int far = 0;
  for (int it = 0; it < NPOINT_; ++it){
    float cx = sx[far*3+0], cy = sx[far*3+1], cz = sx[far*3+2];
    if (tid == 0){
      size_t o = ((size_t)b*NPOINT_ + it)*3;
      out[o+0] = cx; out[o+1] = cy; out[o+2] = cz;
    }
    float bd = -1.0f; int bi = tid*16;
#pragma unroll
    for (int r = 0; r < 16; ++r){
      int i = tid*16 + r;
      float dx = sx[i*3+0]-cx, dy = sx[i*3+1]-cy, dz = sx[i*3+2]-cz;
      float d  = dx*dx + dy*dy + dz*dz;
      float nd = fminf(dist[r], d);
      dist[r] = nd;
      if (nd > bd){ bd = nd; bi = i; }   // first-max semantics (ascending i)
    }
    rd[tid] = bd; ri[tid] = bi;
    __syncthreads();
    for (int s = 128; s > 0; s >>= 1){
      if (tid < s && rd[tid+s] > rd[tid]){ rd[tid] = rd[tid+s]; ri[tid] = ri[tid+s]; }
      __syncthreads();
    }
    far = ri[0];
    __syncthreads();
  }
}

// ================= Ball query + grouping: one wave per centroid =============
__device__ __forceinline__ void write_point(_Float16* __restrict__ A, size_t p,
                                            const float* __restrict__ xyz,
                                            const float* __restrict__ points,
                                            int b, int i, float nx, float ny, float nz){
  const float* q  = xyz    + ((size_t)b*N_ + i)*3;
  const float* pp = points + ((size_t)b*N_ + i)*32;
  float c0 = q[0]-nx, c1 = q[1]-ny, c2 = q[2]-nz;
#pragma unroll 4
  for (int k = 0; k < 64; ++k){
    float v;
    if (k == 0) v = c0; else if (k == 1) v = c1; else if (k == 2) v = c2;
    else if (k < 35) v = pp[k-3];
    else v = 0.0f;                       // zero-pad K 35..63
    A[aswz(p, k, 2)] = (_Float16)v;
  }
}

__global__ void group_kernel(const float* __restrict__ xyz, const float* __restrict__ points,
                             const float* __restrict__ newxyz, _Float16* __restrict__ A){
  int lane = threadIdx.x & 31;
  int w = blockIdx.x * (blockDim.x >> 5) + (threadIdx.x >> 5);
  int b = w >> 10, j = w & 1023;
  const float* c3 = newxyz + ((size_t)(b*NPOINT_ + j))*3;
  float nx = c3[0], ny = c3[1], nz = c3[2];
  size_t pbase = ((size_t)(b*NPOINT_ + j))*NSAMPLE_;
  int cnt = 0, first_i = -1;
  for (int base = 0; base < N_ && cnt < NSAMPLE_; base += 32){
    int i = base + lane;
    const float* q = xyz + ((size_t)b*N_ + i)*3;
    float dx = q[0]-nx, dy = q[1]-ny, dz = q[2]-nz;
    bool pred = (dx*dx + dy*dy + dz*dz) <= RAD2;
    unsigned mask = __builtin_amdgcn_ballot_w32(pred);
    if (first_i < 0 && mask) first_i = base + __builtin_ctz(mask);
    if (pred){
      int slot = cnt + __popc(mask & ((1u << lane) - 1u));
      if (slot < NSAMPLE_) write_point(A, pbase + slot, xyz, points, b, i, nx, ny, nz);
    }
    cnt += __popc(mask);
  }
  if (first_i < 0) first_i = 0;          // can't happen (centroid is in-ball)
  int filled = cnt < NSAMPLE_ ? cnt : NSAMPLE_;
  for (int s = filled + lane; s < NSAMPLE_; s += 32)
    write_point(A, pbase + s, xyz, points, b, first_i, nx, ny, nz);
}

// ================= Weight swizzle (B operand, f32 -> f16) ===================
__global__ void bsw_kernel(const float* __restrict__ w, _Float16* __restrict__ Bsw,
                           int in_ch, int out_ch, int numK){
  int tid = blockIdx.x*blockDim.x + threadIdx.x;
  int total = (out_ch/16)*numK*512;
  if (tid >= total) return;
  int i = tid & 15, lane = (tid >> 4) & 31;
  int q = tid >> 9;
  int kc = q % numK, ntile = q / numK;
  int half = lane >> 4;
  int n = ntile*16 + (lane & 15);
  int k = kc*32 + kmapA(i, half);
  float v = (k < in_ch) ? w[(size_t)n*in_ch + k] : 0.0f;
  Bsw[tid] = (_Float16)v;
}

__global__ void zero_stats(float* s){ s[threadIdx.x] = 0.0f; }

// ====== WMMA GEMM: each wave computes 4 N-tiles (64 cols) per A fragment ====
// wave -> (mtile, ng); n-tiles ng*4 .. ng*4+3; K = 64 (2 chunks of 32)
__global__ void gemm_kernel(const _Float16* __restrict__ A, const _Float16* __restrict__ Bw,
                            const float* __restrict__ bias, _Float16* __restrict__ X,
                            float* __restrict__ sum, float* __restrict__ sumsq,
                            int NG, int numK, int C){
  int lane = threadIdx.x & 31;
  int wave = blockIdx.x * (blockDim.x >> 5) + (threadIdx.x >> 5);
  int ng    = wave % NG;
  int mtile = wave / NG;

  // A fragments for both K chunks, reused by all 4 N-tiles
  v16h a0 = *(const v16h*)(A + (((size_t)mtile*numK + 0)*32 + lane)*16);
  v16h a1 = *(const v16h*)(A + (((size_t)mtile*numK + 1)*32 + lane)*16);

  v8f acc[4];
#pragma unroll
  for (int t = 0; t < 4; ++t){
    int ntile = ng*4 + t;
    v16h b0 = *(const v16h*)(Bw + (((size_t)ntile*numK + 0)*32 + lane)*16);
    v16h b1 = *(const v16h*)(Bw + (((size_t)ntile*numK + 1)*32 + lane)*16);
    v8f c = {};
    c = __builtin_amdgcn_wmma_f32_16x16x32_f16(false, a0, false, b0, (short)0, c,
                                               false, false);
    c = __builtin_amdgcn_wmma_f32_16x16x32_f16(false, a1, false, b1, (short)0, c,
                                               false, false);
    acc[t] = c;
  }

  int half = lane >> 4;
  int r = lane & 15;
#pragma unroll
  for (int t = 0; t < 4; ++t){
    int n = (ng*4 + t)*16 + r;
    float bs = bias[n];
    float ls = 0.0f, lq = 0.0f;
#pragma unroll
    for (int i = 0; i < 8; ++i){
      float v = acc[t][i] + bs;          // D element i -> row mtile*16 + i + 8*half
      size_t p = (size_t)mtile*16 + i + 8*half;
      X[p*(size_t)C + n] = (_Float16)v;
      ls += v; lq += v*v;
    }
    atomicAdd(&sum[n],  ls);
    atomicAdd(&sumsq[n], lq);
  }
}

// ================= BN finalize: fold (mu,var,g,be) into scale/shift =========
__global__ void bn_kernel(const float* __restrict__ sum, const float* __restrict__ sumsq,
                          const float* __restrict__ g, const float* __restrict__ be,
                          float* __restrict__ scale, float* __restrict__ shift, float invP){
  int c = threadIdx.x;
  float mean = sum[c]*invP;
  float var  = sumsq[c]*invP - mean*mean;
  float rstd = rsqrtf(var + EPS_);
  float sc = g[c]*rstd;
  scale[c] = sc;
  shift[c] = be[c] - mean*sc;
}

// ============ BN+ReLU apply; re-swizzle f16 for next layer's A ==============
__global__ void apply_kernel(_Float16* __restrict__ X, const float* __restrict__ scale,
                             const float* __restrict__ shift, _Float16* __restrict__ Anext,
                             int shiftC, int numKnext, int last){
  size_t idx = (size_t)blockIdx.x*blockDim.x + threadIdx.x;
  int c = (int)(idx & ((1u << shiftC) - 1u));
  size_t p = idx >> shiftC;
  float y = (float)X[idx]*scale[c] + shift[c];
  y = fmaxf(y, 0.0f);
  if (last) X[idx] = (_Float16)y;                  // in-place for the max stage
  else      Anext[aswz(p, c, numKnext)] = (_Float16)y;
}

// ================= Max over nsample -> (B,128,NPOINT) =======================
__global__ void max_kernel(const _Float16* __restrict__ X, float* __restrict__ out){
  int tid = blockIdx.x*blockDim.x + threadIdx.x;   // (b*128 + c)*1024 + j
  int j = tid & 1023;
  int c = (tid >> 10) & 127;
  int b = tid >> 17;
  size_t base = ((size_t)(b*NPOINT_ + j))*NSAMPLE_*128 + c;
  float m = -1e30f;
#pragma unroll 8
  for (int s = 0; s < NSAMPLE_; ++s)
    m = fmaxf(m, (float)X[base + (size_t)s*128]);
  out[tid] = m;
}

extern "C" void kernel_launch(void* const* d_in, const int* in_sizes, int n_in,
                              void* d_out, int out_size, void* d_ws, size_t ws_size,
                              hipStream_t stream){
  (void)in_sizes; (void)n_in; (void)out_size; (void)ws_size;
  const float* xyz    = (const float*)d_in[0];
  const float* points = (const float*)d_in[1];
  const float* w[3]  = {(const float*)d_in[2],  (const float*)d_in[6],  (const float*)d_in[10]};
  const float* bb[3] = {(const float*)d_in[3],  (const float*)d_in[7],  (const float*)d_in[11]};
  const float* g[3]  = {(const float*)d_in[4],  (const float*)d_in[8],  (const float*)d_in[12]};
  const float* be[3] = {(const float*)d_in[5],  (const float*)d_in[9],  (const float*)d_in[13]};
  float* out = (float*)d_out;

  char* ws = (char*)d_ws;
  _Float16* Bsw0  = (_Float16*)(ws + 0);
  _Float16* Bsw1  = (_Float16*)(ws + 8192);
  _Float16* Bsw2  = (_Float16*)(ws + 16384);
  float*    stats = (float*)(ws + 32768);             // sum128|sumsq128|scale128|shift128
  _Float16* A     = (_Float16*)(ws + (size_t)(1<<20));
  _Float16* X     = (_Float16*)(ws + (size_t)(1<<20) + (size_t)P_*64*2);
  float *sum = stats, *sumsq = stats+128, *scale = stats+256, *shift = stats+384;

  // new_xyz (output 0) + grouped features (swizzled f16 A, K padded 35->64)
  fps_kernel  <<<B_, 256, 0, stream>>>(xyz, out);
  bsw_kernel  <<<16, 256, 0, stream>>>(w[0], Bsw0, 35, 64, 2);
  bsw_kernel  <<<16, 256, 0, stream>>>(w[1], Bsw1, 64, 64, 2);
  bsw_kernel  <<<32, 256, 0, stream>>>(w[2], Bsw2, 64, 128, 2);
  group_kernel<<<(B_*NPOINT_)/4, 128, 0, stream>>>(xyz, points, out, A);

  // layer 0: 35(->64) -> 64   (NG=1: one wave covers the whole 64-col row)
  zero_stats  <<<1, 512, 0, stream>>>(stats);
  gemm_kernel <<<MT_*1/8, 256, 0, stream>>>(A, Bsw0, bb[0], X, sum, sumsq, 1, 2, 64);
  bn_kernel   <<<1, 64, 0, stream>>>(sum, sumsq, g[0], be[0], scale, shift, 1.0f/P_);
  apply_kernel<<<(P_*64)/256, 256, 0, stream>>>(X, scale, shift, A, 6, 2, 0);

  // layer 1: 64 -> 64
  zero_stats  <<<1, 512, 0, stream>>>(stats);
  gemm_kernel <<<MT_*1/8, 256, 0, stream>>>(A, Bsw1, bb[1], X, sum, sumsq, 1, 2, 64);
  bn_kernel   <<<1, 64, 0, stream>>>(sum, sumsq, g[1], be[1], scale, shift, 1.0f/P_);
  apply_kernel<<<(P_*64)/256, 256, 0, stream>>>(X, scale, shift, A, 6, 2, 0);

  // layer 2: 64 -> 128  (NG=2: two waves per M-tile row)
  zero_stats  <<<1, 512, 0, stream>>>(stats);
  gemm_kernel <<<MT_*2/8, 256, 0, stream>>>(A, Bsw2, bb[2], X, sum, sumsq, 2, 2, 128);
  bn_kernel   <<<1, 128, 0, stream>>>(sum, sumsq, g[2], be[2], scale, shift, 1.0f/P_);
  apply_kernel<<<(P_*128)/256, 256, 0, stream>>>(X, scale, shift, A, 7, 2, 1);

  // max over nsample -> output 1 at offset B*NPOINT*3
  max_kernel  <<<(B_*128*NPOINT_)/256, 256, 0, stream>>>(X, out + (size_t)B_*NPOINT_*3);
}
